// PointNetFeaturePropagation_13503377178960
// MI455X (gfx1250) — matrix operand order repaired and checked by reading, not compiled, not run on hardware
//
#include <hip/hip_runtime.h>

// Problem constants
#define Bb   8
#define Nn   8192
#define Mm   2048
#define C1c  256
#define C2c  512
#define BNt  65536          // B*N
#define INCH 768            // C1 + C2

typedef __attribute__((ext_vector_type(16))) _Float16 v16h;
typedef __attribute__((ext_vector_type(8)))  _Float16 half8;
typedef __attribute__((ext_vector_type(8)))  float    v8f;

#if defined(__has_builtin)
#if __has_builtin(__builtin_amdgcn_tensor_load_to_lds) && \
    __has_builtin(__builtin_amdgcn_s_wait_tensorcnt)
#define USE_TDM 1
#endif
#if __has_builtin(__builtin_amdgcn_sched_barrier)
#define SCHED_FENCE() __builtin_amdgcn_sched_barrier(0)
#endif
#endif
#ifndef SCHED_FENCE
#define SCHED_FENCE()
#endif

#if defined(USE_TDM)
typedef __attribute__((ext_vector_type(4))) unsigned int u32x4;
typedef __attribute__((ext_vector_type(8))) int i32x8;
typedef __attribute__((ext_vector_type(4))) int i32x4;

// Issue one TDM 2D-tile load: 128 rows (Cout dim) x 32 halfs (K dim) from
// row-major (Cout, K) f16 weights into LDS, hardware-padded to a 40-half row
// stride (pad_interval = 16 DWORDs per row, pad_amount = 4 DWORDs = 8 halfs).
__device__ __forceinline__ void tdm_load_tile(const _Float16* gsrc, _Float16* lds_dst,
                                              int K, int Cout) {
    unsigned long long ga = (unsigned long long)(const void*)gsrc;
    unsigned lds = (unsigned)(unsigned long long)(void*)lds_dst;
    u32x4 g0;
    g0.x = 1u;                                            // count=1, user descriptor
    g0.y = lds;                                           // lds_addr
    g0.z = (unsigned)ga;                                  // global_addr[31:0]
    g0.w = (unsigned)((ga >> 32) & 0x01FFFFFFull) | (2u << 30);  // addr[56:32] | type=2
    i32x8 g1;
    g1[0] = (1 << 16)    // data_size = 2 bytes
          | (1 << 20)    // pad_enable
          | (3 << 22)    // pad_interval: 16 DWORDs (one 32-half row)
          | (3 << 25);   // pad_amount: 4 DWORDs (8 halfs) -> stride 40 halfs
    g1[1] = (int)(((unsigned)K & 0xFFFFu) << 16);                       // tensor_dim0[15:0]
    g1[2] = (int)(((unsigned)K >> 16) | (((unsigned)Cout & 0xFFFFu) << 16)); // dim0 hi | dim1 lo
    g1[3] = (int)(((unsigned)Cout >> 16) | (32u << 16));                // dim1 hi | tile_dim0=32
    g1[4] = 128;                                                        // tile_dim1 = 128 rows
    g1[5] = K;                                                          // tensor_dim0_stride lo
    g1[6] = 0;
    g1[7] = 0;
    i32x4 gz = {0, 0, 0, 0};
#if __clang_major__ >= 23
    i32x8 gz8 = {0, 0, 0, 0, 0, 0, 0, 0};
    __builtin_amdgcn_tensor_load_to_lds(g0, g1, gz, gz, gz8, 0);
#else
    __builtin_amdgcn_tensor_load_to_lds(g0, g1, gz, gz, 0);
#endif
}
#endif  // USE_TDM

// ---------------------------------------------------------------------------
// 1) 3-NN search + inverse-distance weights.  One block = one (b, 256-n chunk).
// ---------------------------------------------------------------------------
__global__ __launch_bounds__(256) void knn_kernel(const float* __restrict__ unknown,
                                                  const float* __restrict__ known,
                                                  int* __restrict__ idx3,
                                                  float* __restrict__ w3) {
    __shared__ float kx[Mm], ky[Mm], kz[Mm];
    const int b = blockIdx.x >> 5;        // N/256 = 32 chunks per batch
    const int chunk = blockIdx.x & 31;
    const float* kb = known + (long)b * Mm * 3;
    for (int i = threadIdx.x; i < Mm; i += 256) {
        kx[i] = kb[i * 3 + 0];
        ky[i] = kb[i * 3 + 1];
        kz[i] = kb[i * 3 + 2];
    }
    __syncthreads();
    const int n = chunk * 256 + threadIdx.x;
    const long bn = (long)b * Nn + n;
    const float ux = unknown[bn * 3 + 0];
    const float uy = unknown[bn * 3 + 1];
    const float uz = unknown[bn * 3 + 2];
    float d0 = 3.4e38f, d1 = 3.4e38f, d2 = 3.4e38f;
    int i0 = 0, i1 = 0, i2 = 0;
    for (int m = 0; m < Mm; ++m) {
        float dx = ux - kx[m], dy = uy - ky[m], dz = uz - kz[m];
        float d = dx * dx + dy * dy + dz * dz;
        if (d < d2) {
            if (d < d1) {
                d2 = d1; i2 = i1;
                if (d < d0) { d1 = d0; i1 = i0; d0 = d; i0 = m; }
                else        { d1 = d;  i1 = m; }
            } else { d2 = d; i2 = m; }
        }
    }
    float r0 = 1.f / (d0 + 1e-8f), r1 = 1.f / (d1 + 1e-8f), r2 = 1.f / (d2 + 1e-8f);
    float s = 1.f / (r0 + r1 + r2);
    idx3[bn * 3 + 0] = i0; idx3[bn * 3 + 1] = i1; idx3[bn * 3 + 2] = i2;
    w3[bn * 3 + 0] = r0 * s; w3[bn * 3 + 1] = r1 * s; w3[bn * 3 + 2] = r2 * s;
}

// ---------------------------------------------------------------------------
// 2) f32 -> f16 weight conversion
// ---------------------------------------------------------------------------
__global__ void cvt_f16_kernel(const float* __restrict__ src, _Float16* __restrict__ dst, int n) {
    int i = blockIdx.x * 256 + threadIdx.x;
    if (i < n) dst[i] = (_Float16)src[i];
}

// ---------------------------------------------------------------------------
// 3) points1 (B, C1, N) -> X0[bn][c] f16 for c < 256 (LDS transpose)
//    grid (N/32, C1/32, B), block (32, 8)
// ---------------------------------------------------------------------------
__global__ __launch_bounds__(256) void transpose_p1_kernel(const float* __restrict__ p1,
                                                           _Float16* __restrict__ X0) {
    __shared__ float tile[32][33];
    const int n0 = blockIdx.x * 32, c0 = blockIdx.y * 32, b = blockIdx.z;
    const int tx = threadIdx.x, ty = threadIdx.y;
    const float* src = p1 + ((long)b * C1c + c0) * Nn + n0;
    for (int i = 0; i < 4; ++i)
        tile[ty + i * 8][tx] = src[(long)(ty + i * 8) * Nn + tx];  // tile[c_local][n_local]
    __syncthreads();
    for (int i = 0; i < 4; ++i) {
        int nl = ty + i * 8;
        X0[((long)b * Nn + n0 + nl) * INCH + c0 + tx] = (_Float16)tile[tx][nl];
    }
}

// ---------------------------------------------------------------------------
// 4) weighted-gather interpolation -> X0[bn][256+c2] f16.  Block = 8 bn points.
// ---------------------------------------------------------------------------
__global__ __launch_bounds__(256) void interp_kernel(const float* __restrict__ p2,
                                                     const int* __restrict__ idx3,
                                                     const float* __restrict__ w3,
                                                     _Float16* __restrict__ X0) {
    const long bn0 = (long)blockIdx.x * 8;
    const int t = threadIdx.x;
    for (int k = 0; k < 8; ++k) {
        const long bn = bn0 + k;
        const int b = (int)(bn >> 13);               // N = 8192
        const float* pb = p2 + (long)b * C2c * Mm;
        const int i0 = idx3[bn * 3 + 0], i1 = idx3[bn * 3 + 1], i2 = idx3[bn * 3 + 2];
        const float a0 = w3[bn * 3 + 0], a1 = w3[bn * 3 + 1], a2 = w3[bn * 3 + 2];
        _Float16* dst = X0 + bn * INCH + C1c;
        for (int c = t; c < C2c; c += 256) {
            const float* row = pb + (long)c * Mm;
            float v = a0 * row[i0] + a1 * row[i1] + a2 * row[i2];
            dst[c] = (_Float16)v;
        }
    }
}

// ---------------------------------------------------------------------------
// 5) WMMA GEMM:  Y (BN x Cout, f32)  =  X (BN x K, f16)  @  Wt^T (Cout x K, f16)
//    Block tile: 128 rows(bn) x 128 cols(Cout); 8 waves, each 16 rows x 128 cols.
//    W tiles staged by the Tensor Data Mover (double-buffered); A frags loaded
//    straight from global (X is L2-resident).  grid (Cout/128, BN/128), 256 thr.
// ---------------------------------------------------------------------------
__device__ __forceinline__ void mma_step(const _Float16* __restrict__ sbuf,
                                         const _Float16* __restrict__ xrow,
                                         int k0, int r, int hi, v8f acc[8]) {
    // A fragment (16x32): halves 0..15 = K in [16hi, 16hi+16)
    half8 alo = *(const half8*)(xrow + k0 + 16 * hi);
    half8 ahi = *(const half8*)(xrow + k0 + 16 * hi + 8);
    v16h a = __builtin_shufflevector(alo, ahi, 0, 1, 2, 3, 4, 5, 6, 7,
                                     8, 9, 10, 11, 12, 13, 14, 15);
    // Issue ALL B-fragment LDS loads first...
    v16h bf[8];
#pragma unroll
    for (int j = 0; j < 8; ++j) {
        const _Float16* bp = sbuf + (j * 16 + r) * 40 + 16 * hi;
        half8 blo = *(const half8*)(bp);
        half8 bhi = *(const half8*)(bp + 8);
        bf[j] = __builtin_shufflevector(blo, bhi, 0, 1, 2, 3, 4, 5, 6, 7,
                                        8, 9, 10, 11, 12, 13, 14, 15);
    }
    // ...then fence the scheduler so the 8 WMMAs issue back-to-back behind a
    // single ds-counter wait instead of eight full drains.
    SCHED_FENCE();
#pragma unroll
    for (int j = 0; j < 8; ++j)
        acc[j] = __builtin_amdgcn_wmma_f32_16x16x32_f16(
            false, a, false, bf[j], (short)0, acc[j], false, false);
}

__global__ __launch_bounds__(256) void gemm_wmma_kernel(const _Float16* __restrict__ X,
                                                        const _Float16* __restrict__ Wt,
                                                        float* __restrict__ Y,
                                                        int K, int Cout) {
    __shared__ __align__(16) _Float16 sW[2][128 * 40];   // 2 x (128 cols x 32 k, stride 40)
    const int t = threadIdx.x;
    const int lane = t & 31, w = t >> 5;
    const int col0 = blockIdx.x * 128;
    const long row0 = (long)blockIdx.y * 128;
    const int r = lane & 15, hi = lane >> 4;
    const _Float16* xrow = X + (row0 + w * 16 + r) * (long)K;

    v8f acc[8];
    const v8f zz = {0.f, 0.f, 0.f, 0.f, 0.f, 0.f, 0.f, 0.f};
#pragma unroll
    for (int j = 0; j < 8; ++j) acc[j] = zz;

    const int steps = K / 32;

#if defined(USE_TDM)
    const _Float16* wtile0 = Wt + (long)col0 * K;
    if (w == 0) tdm_load_tile(wtile0, &sW[0][0], K, Cout);
    for (int s = 0; s < steps; ++s) {
        const int cur = s & 1;
        const int k0 = s * 32;
        if (w == 0) __builtin_amdgcn_s_wait_tensorcnt(0);   // buffer[cur] filled
        __syncthreads();                                    // visible to all waves;
                                                            // alt buffer's readers done
        if (w == 0 && s + 1 < steps)
            tdm_load_tile(wtile0 + (k0 + 32), &sW[1 - cur][0], K, Cout);
        if (k0 + 64 < K) __builtin_prefetch(xrow + k0 + 64, 0, 0);
        mma_step(&sW[cur][0], xrow, k0, r, hi, acc);
    }
#else
    // Fallback: cooperative load + ds_store staging (single buffer)
    const int scol = t >> 1, sseg = t & 1;
    const _Float16* wsrc = Wt + (long)(col0 + scol) * K + sseg * 16;
    _Float16* sdst = &sW[0][0] + scol * 40 + sseg * 16;
    for (int s = 0; s < steps; ++s) {
        const int k0 = s * 32;
        half8 w0 = *(const half8*)(wsrc + k0);
        half8 w1 = *(const half8*)(wsrc + k0 + 8);
        if (k0 + 64 < K) __builtin_prefetch(xrow + k0 + 64, 0, 0);
        __syncthreads();
        *(half8*)(sdst)     = w0;
        *(half8*)(sdst + 8) = w1;
        __syncthreads();
        mma_step(&sW[0][0], xrow, k0, r, hi, acc);
    }
#endif

    // D layout: VGPR v -> row (v + 8*hi), lane&15 -> col
#pragma unroll
    for (int j = 0; j < 8; ++j) {
        const int c = col0 + j * 16 + r;
#pragma unroll
        for (int v = 0; v < 8; ++v) {
            const long row = row0 + w * 16 + v + 8 * hi;
            Y[row * Cout + c] = acc[j][v];
        }
    }
}

// ---------------------------------------------------------------------------
// 6) per-channel sum / sumsq over columns of Y (BN x Cout).  grid (Cout/256, 64)
// ---------------------------------------------------------------------------
__global__ void zero_kernel(float* __restrict__ p, int n) {
    int i = blockIdx.x * 256 + threadIdx.x;
    if (i < n) p[i] = 0.f;
}

__global__ __launch_bounds__(256) void stats_kernel(const float* __restrict__ Y,
                                                    float* __restrict__ sum,
                                                    float* __restrict__ sumsq,
                                                    int Cout) {
    const int o = blockIdx.x * 256 + threadIdx.x;
    const long r0 = (long)blockIdx.y * 1024;
    const float* p = Y + r0 * Cout + o;
    float s = 0.f, q = 0.f;
    for (int rr = 0; rr < 1024; ++rr) {
        float v = p[(long)rr * Cout];
        s += v;
        q += v * v;
    }
    atomicAdd(&sum[o], s);
    atomicAdd(&sumsq[o], q);
}

__global__ void finalize_kernel(const float* __restrict__ sum, const float* __restrict__ sumsq,
                                const float* __restrict__ gamma, const float* __restrict__ beta,
                                float* __restrict__ scale, float* __restrict__ shift, int Cout) {
    int o = blockIdx.x * 256 + threadIdx.x;
    if (o < Cout) {
        const float inv = 1.0f / (float)BNt;
        float mu = sum[o] * inv;
        float var = sumsq[o] * inv - mu * mu;
        float sc = gamma[o] * rsqrtf(var + 1e-5f);
        scale[o] = sc;
        shift[o] = beta[o] - mu * sc;
    }
}

// ---------------------------------------------------------------------------
// 7) BN + ReLU -> f16 activations (same (BN, Cout) layout).  Cout power of 2.
// ---------------------------------------------------------------------------
__global__ void apply_relu_f16_kernel(const float* __restrict__ Y,
                                      const float* __restrict__ scale,
                                      const float* __restrict__ shift,
                                      _Float16* __restrict__ Xout, int Cout) {
    long i = (long)blockIdx.x * 256 + threadIdx.x;
    int o = (int)(i & (long)(Cout - 1));
    float v = Y[i] * scale[o] + shift[o];
    Xout[i] = (_Float16)(v > 0.f ? v : 0.f);
}

// ---------------------------------------------------------------------------
// 8) BN + ReLU + transpose to (B, 256, N) f32 output.
//    grid (N/32, 256/32, B), block (32, 8)
// ---------------------------------------------------------------------------
__global__ __launch_bounds__(256) void apply_out_kernel(const float* __restrict__ Y,
                                                        const float* __restrict__ scale,
                                                        const float* __restrict__ shift,
                                                        float* __restrict__ out) {
    __shared__ float tile[32][33];
    const int n0 = blockIdx.x * 32, c0 = blockIdx.y * 32, b = blockIdx.z;
    const int tx = threadIdx.x, ty = threadIdx.y;
    for (int i = 0; i < 4; ++i) {
        int nl = ty + i * 8;
        long bn = (long)b * Nn + n0 + nl;
        int c = c0 + tx;
        float v = Y[bn * 256 + c] * scale[c] + shift[c];
        tile[nl][tx] = v > 0.f ? v : 0.f;     // tile[n_local][c_local]
    }
    __syncthreads();
    for (int i = 0; i < 4; ++i) {
        int cl = ty + i * 8;
        out[((long)b * 256 + c0 + cl) * Nn + n0 + tx] = tile[tx][cl];
    }
}

// ---------------------------------------------------------------------------
extern "C" void kernel_launch(void* const* d_in, const int* in_sizes, int n_in,
                              void* d_out, int out_size, void* d_ws, size_t ws_size,
                              hipStream_t stream) {
    const float* unknown = (const float*)d_in[0];
    const float* known   = (const float*)d_in[1];
    const float* points1 = (const float*)d_in[2];
    const float* points2 = (const float*)d_in[3];
    const float* W1      = (const float*)d_in[4];
    // d_in[5] = b1  (cancels through training-mode BN)
    const float* gamma1  = (const float*)d_in[6];
    const float* beta1   = (const float*)d_in[7];
    const float* W2      = (const float*)d_in[8];
    // d_in[9] = b2  (cancels through training-mode BN)
    const float* gamma2  = (const float*)d_in[10];
    const float* beta2   = (const float*)d_in[11];
    float* out = (float*)d_out;

    char* ws = (char*)d_ws;
    size_t off = 0;
    auto alloc = [&](size_t bytes) -> void* {
        void* p = ws + off;
        off = (off + bytes + 255) & ~(size_t)255;
        return p;
    };
    int*      idx3  = (int*)alloc((size_t)BNt * 3 * sizeof(int));
    float*    w3    = (float*)alloc((size_t)BNt * 3 * sizeof(float));
    _Float16* W1h   = (_Float16*)alloc((size_t)512 * INCH * sizeof(_Float16));
    _Float16* W2h   = (_Float16*)alloc((size_t)256 * 512 * sizeof(_Float16));
    float*    s_sum = (float*)alloc(512 * sizeof(float));
    float*    s_sq  = (float*)alloc(512 * sizeof(float));
    float*    s_sc  = (float*)alloc(512 * sizeof(float));
    float*    s_sh  = (float*)alloc(512 * sizeof(float));
    _Float16* X0    = (_Float16*)alloc((size_t)BNt * INCH * sizeof(_Float16)); // 96 MB
    float*    Y     = (float*)alloc((size_t)BNt * 512 * sizeof(float));        // 128 MB (reused)
    _Float16* X1    = X0;   // reuse: X0 dead once GEMM1 finished; X1 needs 64 MB <= 96 MB

    // --- stage 1: kNN + weights ---
    knn_kernel<<<Bb * (Nn / 256), 256, 0, stream>>>(unknown, known, idx3, w3);

    // --- stage 2: convert weights to f16 ---
    cvt_f16_kernel<<<(512 * INCH + 255) / 256, 256, 0, stream>>>(W1, W1h, 512 * INCH);
    cvt_f16_kernel<<<(256 * 512 + 255) / 256, 256, 0, stream>>>(W2, W2h, 256 * 512);

    // --- stage 3: build X0 = [points1^T | interp] in f16, (BN x 768) ---
    transpose_p1_kernel<<<dim3(Nn / 32, C1c / 32, Bb), dim3(32, 8), 0, stream>>>(points1, X0);
    interp_kernel<<<BNt / 8, 256, 0, stream>>>(points2, idx3, w3, X0);

    // --- layer 1: GEMM + BN + ReLU ---
    gemm_wmma_kernel<<<dim3(512 / 128, BNt / 128), 256, 0, stream>>>(X0, W1h, Y, INCH, 512);
    zero_kernel<<<4, 256, 0, stream>>>(s_sum, 512);
    zero_kernel<<<4, 256, 0, stream>>>(s_sq, 512);
    stats_kernel<<<dim3(512 / 256, BNt / 1024), 256, 0, stream>>>(Y, s_sum, s_sq, 512);
    finalize_kernel<<<2, 256, 0, stream>>>(s_sum, s_sq, gamma1, beta1, s_sc, s_sh, 512);
    apply_relu_f16_kernel<<<(int)(((long)BNt * 512) / 256), 256, 0, stream>>>(Y, s_sc, s_sh, X1, 512);

    // --- layer 2: GEMM + BN + ReLU + transpose out ---
    gemm_wmma_kernel<<<dim3(256 / 128, BNt / 128), 256, 0, stream>>>(X1, W2h, Y, 512, 256);
    zero_kernel<<<2, 256, 0, stream>>>(s_sum, 256);
    zero_kernel<<<2, 256, 0, stream>>>(s_sq, 256);
    stats_kernel<<<dim3(256 / 256, BNt / 1024), 256, 0, stream>>>(Y, s_sum, s_sq, 256);
    finalize_kernel<<<1, 256, 0, stream>>>(s_sum, s_sq, gamma2, beta2, s_sc, s_sh, 256);
    apply_out_kernel<<<dim3(Nn / 32, 256 / 32, Bb), dim3(32, 8), 0, stream>>>(Y, s_sc, s_sh, out);
}
